// MultiHeadedAttention_10290741641957
// MI455X (gfx1250) — matrix operand order repaired and checked by reading, not compiled, not run on hardware
//
#include <hip/hip_runtime.h>
#include <hip/hip_bf16.h>

#define B_  16
#define S_  4096
#define DQ_ 1024
#define DM_ 512
#define DO_ 1024
#define H_  16
#define D_  64

typedef __attribute__((ext_vector_type(16))) __bf16 v16bf;
typedef __attribute__((ext_vector_type(8)))  __bf16 v8bf;
typedef __attribute__((ext_vector_type(8)))  float  v8f;

// ---------------- Kernel 1: q = (source @ Wq) * (1/sqrt(D)) ----------------
__global__ __launch_bounds__(256) void k_q(const float* __restrict__ src,
                                           const float* __restrict__ Wq,
                                           float* __restrict__ q) {
  int idx = blockIdx.x * 256 + threadIdx.x;      // B*DO = 16384
  int b = idx >> 10, o = idx & (DO_ - 1);
  const float* s = src + (size_t)b * DQ_;
  float acc = 0.f;
  for (int k = 0; k < DQ_; ++k) acc += s[k] * Wq[(size_t)k * DO_ + o];
  q[idx] = acc * 0.125f;                          // 1/sqrt(64)
}

// ---------------- Kernel 2: qk[b,h,m] = sum_d q[b,h*64+d] * Wk[m, h*64+d] ----
__global__ __launch_bounds__(256) void k_qk(const float* __restrict__ q,
                                            const float* __restrict__ Wk,
                                            float* __restrict__ qk) {
  int idx = blockIdx.x * 256 + threadIdx.x;      // B*H*DM = 131072
  int m = idx & (DM_ - 1);
  int h = (idx >> 9) & (H_ - 1);
  int b = idx >> 13;
  const float* qrow = q + (size_t)b * DO_ + h * D_;
  const float* wrow = Wk + (size_t)m * DO_ + h * D_;
  float acc = 0.f;
#pragma unroll
  for (int d = 0; d < D_; ++d) acc += qrow[d] * wrow[d];
  qk[idx] = acc;                                  // layout ((b*H+h)*DM+m)
}

// ---------------- Kernel 3: scores[b,h,s] = qk[b,h,:] . mem[b,s,:] (WMMA) ----
// grid = (S/128, B), 256 threads = 8 waves, each wave one 16x16 (head x seq) tile
__global__ __launch_bounds__(256) void k_scores(const float* __restrict__ qk,
                                                const float* __restrict__ mem,
                                                const unsigned char* __restrict__ mask,
                                                float* __restrict__ scores) {
  __shared__ __bf16 qk_lds[H_ * DM_];            // 16 KB, bf16 A operand
  int b = blockIdx.y;
  for (int i = threadIdx.x; i < H_ * DM_; i += 256)
    qk_lds[i] = (__bf16)qk[(size_t)b * H_ * DM_ + i];
  __syncthreads();

  int wave = threadIdx.x >> 5;
  int lane = threadIdx.x & 31;
  int half = lane >> 4;                          // 0: K-lo lanes, 1: K-hi lanes
  int nn   = lane & 15;                          // A: head index / B: seq index
  int s0   = blockIdx.x * 128 + wave * 16;
  const float* Mrow = mem + ((size_t)b * S_ + (size_t)(s0 + nn)) * DM_;

  v8f c = {};
  for (int kk = 0; kk < DM_ / 32; ++kk) {
    // A frag (16x32 bf16): lane<16 -> K {0..7,16..23}; lane>=16 -> {8..15,24..31}
    const v8bf* ap = (const v8bf*)(qk_lds + nn * DM_ + kk * 32 + half * 8);
    v8bf alo = ap[0];
    v8bf ahi = ap[2];                            // +16 bf16 elements
    v16bf a;
#pragma unroll
    for (int e = 0; e < 8; ++e) { a[e] = alo[e]; a[8 + e] = ahi[e]; }

    // B frag (32x16 bf16): lane j holds column N=j (seq row), K {0..7,16..23} / {8..15,24..31}
    const float4* bp = (const float4*)(Mrow + kk * 32 + half * 8);
    float4 f0 = bp[0], f1 = bp[1], f2 = bp[4], f3 = bp[5];
    v16bf bb;
    bb[0]=(__bf16)f0.x; bb[1]=(__bf16)f0.y; bb[2]=(__bf16)f0.z; bb[3]=(__bf16)f0.w;
    bb[4]=(__bf16)f1.x; bb[5]=(__bf16)f1.y; bb[6]=(__bf16)f1.z; bb[7]=(__bf16)f1.w;
    bb[8]=(__bf16)f2.x; bb[9]=(__bf16)f2.y; bb[10]=(__bf16)f2.z; bb[11]=(__bf16)f2.w;
    bb[12]=(__bf16)f3.x; bb[13]=(__bf16)f3.y; bb[14]=(__bf16)f3.z; bb[15]=(__bf16)f3.w;

    c = __builtin_amdgcn_wmma_f32_16x16x32_bf16(false, a, false, bb,
                                                (short)0, c, false, false);
  }

  // C layout: VGPR r, lanes<16 -> head r, lanes>=16 -> head r+8; N = lane&15
  int s = s0 + nn;
  unsigned char mk = mask[(size_t)b * S_ + s];
#pragma unroll
  for (int r = 0; r < 8; ++r) {
    int head = r + half * 8;
    float v = mk ? -1e18f : c[r];
    scores[((size_t)b * H_ + head) * S_ + s] = v;
  }
}

// ---------------- Kernel 4: softmax over s, write attn (out + ws) -----------
__global__ __launch_bounds__(256) void k_softmax(float* __restrict__ scores,
                                                 float* __restrict__ attn_out) {
  int row = blockIdx.x;                          // b*H + h  (256 rows)
  float* sp = scores + (size_t)row * S_;
  __shared__ float red[256];
  float vals[16];
  float lmax = -3.4e38f;
#pragma unroll
  for (int i = 0; i < 16; ++i) {
    vals[i] = sp[threadIdx.x + i * 256];
    lmax = fmaxf(lmax, vals[i]);
  }
  red[threadIdx.x] = lmax; __syncthreads();
  for (int off = 128; off > 0; off >>= 1) {
    if (threadIdx.x < off) red[threadIdx.x] = fmaxf(red[threadIdx.x], red[threadIdx.x + off]);
    __syncthreads();
  }
  float m = red[0]; __syncthreads();
  float lsum = 0.f;
#pragma unroll
  for (int i = 0; i < 16; ++i) { vals[i] = __expf(vals[i] - m); lsum += vals[i]; }
  red[threadIdx.x] = lsum; __syncthreads();
  for (int off = 128; off > 0; off >>= 1) {
    if (threadIdx.x < off) red[threadIdx.x] += red[threadIdx.x + off];
    __syncthreads();
  }
  float inv = 1.0f / red[0];
#pragma unroll
  for (int i = 0; i < 16; ++i) {
    float p = vals[i] * inv;
    sp[threadIdx.x + i * 256] = p;               // reuse ws for kernel 5
    attn_out[(size_t)row * S_ + threadIdx.x + i * 256] = p;
  }
}

// ---------------- Kernel 5: ctxm[b,h,m] = sum_s attn[b,h,s]*mem[b,s,m] (WMMA)
// grid = (DM/16, B): one 16-wide m-tile per WG; 8 waves split K=S into 512-slices.
// Each wave double-buffers its 32x16 f32 chunk into LDS with async loads.
__global__ __launch_bounds__(256) void k_ctx(const float* __restrict__ attn,
                                             const float* __restrict__ mem,
                                             float* __restrict__ ctxm) {
  __shared__ float stage[2 * 8 * 512];           // 2 bufs x 8 waves x (32s x 16m) f32
  __shared__ float part[8 * 8 * 32];             // per-wave partial C tiles

  int b    = blockIdx.y;
  int m0   = blockIdx.x * 16;
  int wave = threadIdx.x >> 5;
  int lane = threadIdx.x & 31;
  int half = lane >> 4;
  int nn   = lane & 15;

  const float* mbase = mem + (size_t)b * S_ * DM_;
  const float* arow  = attn + ((size_t)b * H_ + nn) * S_ + wave * 512; // head = nn

  // Async-issue chunk kk of this wave's K-slice into stage buffer `buf`.
  auto issue = [&](int kk, int buf) {
    int s_base = wave * 512 + kk * 32;
    unsigned lbase = (unsigned)(uintptr_t)(&stage[(buf * 8 + wave) * 512]);
#pragma unroll
    for (int i = 0; i < 4; ++i) {
      int elem = i * 32 + lane;                  // elem = row*4 + quarter
      int row  = elem >> 2;
      int q    = elem & 3;
      unsigned goff = (unsigned)(((s_base + row) * DM_ + m0) * 4 + q * 16);
      unsigned loff = lbase + (unsigned)elem * 16u;
      asm volatile("global_load_async_to_lds_b128 %0, %1, %2"
                   :: "v"(loff), "v"(goff), "s"(mbase) : "memory");
    }
  };

  issue(0, 0);
  v8f c = {};
  for (int kk = 0; kk < 16; ++kk) {
    int buf = kk & 1;
    if (kk + 1 < 16) {
      issue(kk + 1, buf ^ 1);
      asm volatile("s_wait_asynccnt 0x4" ::: "memory");  // current buf done
    } else {
      asm volatile("s_wait_asynccnt 0x0" ::: "memory");
    }
    const float* sv = &stage[(buf * 8 + wave) * 512];    // [32 s][16 m] f32

    // A frag from attn (row-contiguous f32 -> bf16)
    const float4* ap4 = (const float4*)(arow + kk * 32 + half * 8);
    float4 a0 = ap4[0], a1 = ap4[1];             // K base .. base+7
    float4 a2 = ap4[4], a3 = ap4[5];             // K base+16 .. base+23
    v16bf a;
    a[0]=(__bf16)a0.x; a[1]=(__bf16)a0.y; a[2]=(__bf16)a0.z; a[3]=(__bf16)a0.w;
    a[4]=(__bf16)a1.x; a[5]=(__bf16)a1.y; a[6]=(__bf16)a1.z; a[7]=(__bf16)a1.w;
    a[8]=(__bf16)a2.x; a[9]=(__bf16)a2.y; a[10]=(__bf16)a2.z; a[11]=(__bf16)a2.w;
    a[12]=(__bf16)a3.x; a[13]=(__bf16)a3.y; a[14]=(__bf16)a3.z; a[15]=(__bf16)a3.w;

    // B frag (32x16): lane j -> column j, K = half*8 + e + (e&8); conflict-free LDS reads
    v16bf bb;
#pragma unroll
    for (int e = 0; e < 16; ++e)
      bb[e] = (__bf16)sv[(half * 8 + e + (e & 8)) * 16 + nn];

    c = __builtin_amdgcn_wmma_f32_16x16x32_bf16(false, a, false, bb,
                                                (short)0, c, false, false);
  }

  // Reduce the 8 per-wave partials for this (16h x 16m) tile.
#pragma unroll
  for (int r = 0; r < 8; ++r)
    part[(wave * 8 + r) * 32 + lane] = c[r];
  __syncthreads();

  int r2 = threadIdx.x >> 5;                     // slot (r2, l2) per thread
  int l2 = threadIdx.x & 31;
  float acc = 0.f;
#pragma unroll
  for (int w = 0; w < 8; ++w) acc += part[(w * 8 + r2) * 32 + l2];
  int head = r2 + (l2 >> 4) * 8;
  int mcol = m0 + (l2 & 15);
  ctxm[((size_t)b * H_ + head) * DM_ + mcol] = acc;
}

// ---------------- Kernel 6: context[b, h*64+dd] = ctxm[b,h,:] @ Wv[:, h*64+dd]
__global__ __launch_bounds__(256) void k_out(const float* __restrict__ ctxm,
                                             const float* __restrict__ Wv,
                                             float* __restrict__ out) {
  int idx = blockIdx.x * 256 + threadIdx.x;      // B*DO = 16384
  int b = idx >> 10;
  int o = idx & (DO_ - 1);
  int h = o >> 6;
  const float* crow = ctxm + ((size_t)b * H_ + h) * DM_;
  float acc = 0.f;
  for (int m = 0; m < DM_; ++m) acc += crow[m] * Wv[(size_t)m * DO_ + o];
  out[idx] = acc;
}

extern "C" void kernel_launch(void* const* d_in, const int* in_sizes, int n_in,
                              void* d_out, int out_size, void* d_ws, size_t ws_size,
                              hipStream_t stream) {
  const float* source = (const float*)d_in[0];
  const float* membank = (const float*)d_in[1];
  const float* Wq = (const float*)d_in[2];
  const float* Wk = (const float*)d_in[3];
  const float* Wv = (const float*)d_in[4];
  const unsigned char* mask = (const unsigned char*)d_in[5];

  float* out_ctx  = (float*)d_out;                       // [B, H*D] = 16384
  float* out_attn = (float*)d_out + (size_t)B_ * DO_;    // [B, H, S]

  float* ws     = (float*)d_ws;
  float* w_q    = ws;                                    // 16384
  float* w_qk   = w_q + (size_t)B_ * DO_;                // 131072
  float* w_sc   = w_qk + (size_t)B_ * H_ * DM_;          // 1048576 (scores->attn)
  float* w_ctxm = w_sc + (size_t)B_ * H_ * S_;           // 131072

  k_q      <<<dim3(B_ * DO_ / 256), 256, 0, stream>>>(source, Wq, w_q);
  k_qk     <<<dim3(B_ * H_ * DM_ / 256), 256, 0, stream>>>(w_q, Wk, w_qk);
  k_scores <<<dim3(S_ / 128, B_), 256, 0, stream>>>(w_qk, membank, mask, w_sc);
  k_softmax<<<dim3(B_ * H_), 256, 0, stream>>>(w_sc, out_attn);
  k_ctx    <<<dim3(DM_ / 16, B_), 256, 0, stream>>>(w_sc, membank, w_ctxm);
  k_out    <<<dim3(B_ * DO_ / 256), 256, 0, stream>>>(w_ctxm, Wv, out_ctx);
}
